// NKATSelfAttention_2121713844863
// MI455X (gfx1250) — matrix operand order repaired and checked by reading, not compiled, not run on hardware
//
#include <hip/hip_runtime.h>
#include <hip/hip_bf16.h>

// ---------------------------------------------------------------------------
// NKAT self-attention with top-k + top-p filtering, CDNA5 (gfx1250) WMMA path.
//
// B=2, N=2048, C=768, H=12, D=64, top_k=64, TOP_P=0.9
//
// Stages (all f32 accumulation through v_wmma_f32_16x16x32_f16):
//   0) convert x, qkv_w, proj_w to f16 (workspace)
//   1) QKV GEMM  [4096x768] x [768x2304]^T -> q[B,H,N,D], k[B,H,N,D], vT[B,H,D,N] (f16)
//   2) fused attention per (b,h,16-row tile):
//        S = Q K^T * scale  (WMMA, into 128KB dynamic LDS, f32)
//        per row: bisect k-th-largest value, bisect nucleus threshold,
//                 masked softmax written back to LDS
//        O = P V            (WMMA, A from LDS converted f32->f16)
//   3) out projection GEMM + bias -> d_out (f32)
// ---------------------------------------------------------------------------

typedef __attribute__((ext_vector_type(16))) _Float16 v16h;
typedef __attribute__((ext_vector_type(8)))  _Float16 v8h;
typedef __attribute__((ext_vector_type(8)))  float    v8f;

#define WMMA_F16F32(a, b, c) \
  __builtin_amdgcn_wmma_f32_16x16x32_f16(false, (a), false, (b), (short)0, (c), false, false)

static __device__ __forceinline__ v16h load_a16(const _Float16* __restrict__ row, int base) {
  // A-matrix 16x32 f16 layout: lane holds M=lane&15; halves 0..7 -> K=base..base+7,
  // halves 8..15 -> K=base+16..base+23 (base already includes 8*(lane>>4)).
  const v8h lo = *reinterpret_cast<const v8h*>(row + base);
  const v8h hi = *reinterpret_cast<const v8h*>(row + base + 16);
  v16h a;
#pragma unroll
  for (int i = 0; i < 8; ++i) { a[i] = lo[i]; a[i + 8] = hi[i]; }
  return a;
}

static __device__ __forceinline__ v16h load_b16(const _Float16* __restrict__ row, int off) {
  // B-matrix 32x16 f16 layout: lane holds column N=lane&15; halves 0..7 -> K=off..off+7,
  // halves 8..15 -> K=off+16..off+23 (off already includes 8*(lane>>4)).
  const v8h lo = *reinterpret_cast<const v8h*>(row + off);
  const v8h hi = *reinterpret_cast<const v8h*>(row + off + 16);
  v16h b;
#pragma unroll
  for (int i = 0; i < 8; ++i) { b[i] = lo[i]; b[i + 8] = hi[i]; }
  return b;
}

static __device__ __forceinline__ float wred_max(float x) {
#pragma unroll
  for (int o = 16; o > 0; o >>= 1) x = fmaxf(x, __shfl_xor(x, o));
  return x;
}
static __device__ __forceinline__ float wred_min(float x) {
#pragma unroll
  for (int o = 16; o > 0; o >>= 1) x = fminf(x, __shfl_xor(x, o));
  return x;
}
static __device__ __forceinline__ float wred_addf(float x) {
#pragma unroll
  for (int o = 16; o > 0; o >>= 1) x += __shfl_xor(x, o);
  return x;
}
static __device__ __forceinline__ int wred_addi(int x) {
#pragma unroll
  for (int o = 16; o > 0; o >>= 1) x += __shfl_xor(x, o);
  return x;
}

// ---------------------------------------------------------------------------
__global__ void cvt_f32_to_f16(const float* __restrict__ src, _Float16* __restrict__ dst, int n) {
  int i = blockIdx.x * blockDim.x + threadIdx.x;
  if (i < n) dst[i] = (_Float16)src[i];
}

// ---------------------------------------------------------------------------
// Stage 1: QKV = x @ qkv_w^T, scattered to q[B,H,N,D], k[B,H,N,D], vT[B,H,D,N].
// One 16x16 output tile per wave, 4 waves per block. M=4096, Ncols=2304, K=768.
__global__ void qkv_gemm(const _Float16* __restrict__ xh,
                         const _Float16* __restrict__ wh,
                         _Float16* __restrict__ qh,
                         _Float16* __restrict__ kh,
                         _Float16* __restrict__ vth) {
  const int lane = threadIdx.x & 31;
  const int wave = threadIdx.x >> 5;
  const int tile = blockIdx.x * 4 + wave;     // 256*144 = 36864 tiles
  const int NT = 2304 / 16;                   // 144
  const int mt = tile / NT;
  const int nt = tile % NT;
  const int asel = (lane >> 4) << 3;          // +8 halves for upper half-wave
  const int bsel = (lane >> 4) << 3;          // +8 halves for upper half-wave

  const _Float16* arow = xh + (mt * 16 + (lane & 15)) * 768;
  const _Float16* brow = wh + (nt * 16 + (lane & 15)) * 768;

  v8f c = {};
  for (int kc = 0; kc < 768; kc += 32) {
    v16h a = load_a16(arow, kc + asel);
    v16h b = load_b16(brow, kc + bsel);
    c = WMMA_F16F32(a, b, c);
  }

#pragma unroll
  for (int g = 0; g < 8; ++g) {
    const int r  = mt * 16 + g + ((lane >> 4) << 3);  // global row (b*N + n)
    const int cc = nt * 16 + (lane & 15);             // column in [0, 2304)
    const int bb = r >> 11, n = r & 2047;
    const int three = cc / 768;
    const int rem = cc - three * 768;
    const int hh = rem >> 6, dd = rem & 63;
    const _Float16 val = (_Float16)c[g];
    if (three == 0)      qh[((bb * 12 + hh) * 2048 + n) * 64 + dd] = val;
    else if (three == 1) kh[((bb * 12 + hh) * 2048 + n) * 64 + dd] = val;
    else                 vth[((bb * 12 + hh) * 64 + dd) * 2048 + n] = val;
  }
}

// ---------------------------------------------------------------------------
// Stage 2: fused attention. One block (128 thr / 4 waves) per (b,h, 16-row tile).
// Dynamic LDS: S[16][2048] f32 = 131072 bytes.
__global__ void attn_fused(const _Float16* __restrict__ qh,
                           const _Float16* __restrict__ kh,
                           const _Float16* __restrict__ vth,
                           _Float16* __restrict__ aoh,
                           const float* __restrict__ temperature,
                           const int* __restrict__ topk_p) {
  extern __shared__ char smem[];
  float* S = (float*)smem;                     // [16][2048]

  const int lane = threadIdx.x & 31;
  const int wave = threadIdx.x >> 5;
  const int bh = blockIdx.x >> 7;              // 128 row-tiles per (b,h)
  const int nt = blockIdx.x & 127;
  const int b = bh / 12, h = bh % 12;
  const float scale = 0.125f * temperature[0]; // D^-0.5 * temperature
  const int topk = topk_p[0];

  const _Float16* Qbase = qh  + (size_t)bh * 2048 * 64;
  const _Float16* Kbase = kh  + (size_t)bh * 2048 * 64;
  const _Float16* Vt    = vth + (size_t)bh * 64 * 2048;

  const int asel = (lane >> 4) << 3;
  const int bsel = (lane >> 4) << 3;

  // ---- S = Q K^T * scale -------------------------------------------------
  const _Float16* qrow = Qbase + (nt * 16 + (lane & 15)) * 64;
  const v16h a0 = load_a16(qrow, 0 + asel);
  const v16h a1 = load_a16(qrow, 32 + asel);

  for (int ct = wave; ct < 128; ct += 4) {
    const _Float16* krow = Kbase + (ct * 16 + (lane & 15)) * 64;
    v8f c = {};
    c = WMMA_F16F32(a0, load_b16(krow, 0 + bsel), c);
    c = WMMA_F16F32(a1, load_b16(krow, 32 + bsel), c);
#pragma unroll
    for (int g = 0; g < 8; ++g) {
      const int r   = g + ((lane >> 4) << 3);
      const int col = ct * 16 + (lane & 15);
      S[r * 2048 + col] = c[g] * scale;
    }
  }
  __syncthreads();

  // ---- per-row top-k / top-p filtering + softmax -------------------------
  for (int rr = 0; rr < 4; ++rr) {
    const int row = wave * 4 + rr;
    float* Srow = S + row * 2048;

    float v[64];
#pragma unroll
    for (int j = 0; j < 64; ++j) v[j] = Srow[lane + 32 * j];

    float rmax = -3.402823e38f, rmin = 3.402823e38f;
#pragma unroll
    for (int j = 0; j < 64; ++j) { rmax = fmaxf(rmax, v[j]); rmin = fminf(rmin, v[j]); }
    rmax = wred_max(rmax);
    rmin = wred_min(rmin);

    // Bisect the k-th largest value: keep set = {s >= vk} (ties kept, as in ref).
    float lo = rmin, hi = rmax;
    for (int it = 0; it < 24; ++it) {
      const float mid = 0.5f * (lo + hi);
      int cnt = 0;
#pragma unroll
      for (int j = 0; j < 64; ++j) cnt += (v[j] >= mid) ? 1 : 0;
      cnt = wred_addi(cnt);
      if (cnt >= topk) lo = mid; else hi = mid;
    }
    const float vk = lo;

    // Softmax mass of the top-k survivors.
    float Z = 0.f;
#pragma unroll
    for (int j = 0; j < 64; ++j) Z += (v[j] >= vk) ? __expf(v[j] - rmax) : 0.f;
    Z = wred_addf(Z);
    const float target = 0.9f * Z;

    // Bisect nucleus threshold t*: element kept iff mass of strictly-greater
    // survivors <= 0.9*Z  (monotone in the value axis).
    float plo = vk, phi = rmax;
    for (int it = 0; it < 24; ++it) {
      const float mid = 0.5f * (plo + phi);
      float F = 0.f;
#pragma unroll
      for (int j = 0; j < 64; ++j)
        F += (v[j] >= vk && v[j] > mid) ? __expf(v[j] - rmax) : 0.f;
      F = wred_addf(F);
      if (F <= target) phi = mid; else plo = mid;
    }
    const float tp = phi;

    float Zp = 0.f;
#pragma unroll
    for (int j = 0; j < 64; ++j) {
      const float p = (v[j] >= vk && v[j] >= tp) ? __expf(v[j] - rmax) : 0.f;
      v[j] = p;
      Zp += p;
    }
    Zp = wred_addf(Zp);
    const float invZ = 1.0f / Zp;
#pragma unroll
    for (int j = 0; j < 64; ++j) Srow[lane + 32 * j] = v[j] * invZ;
  }
  __syncthreads();

  // ---- O = P V  (P from LDS f32 -> f16 on the fly; V stored transposed) --
  const int dt = wave * 16;
  const _Float16* vrow = Vt + (dt + (lane & 15)) * 2048;
  const float* Sa = S + (lane & 15) * 2048;
  v8f acc = {};
  for (int kc = 0; kc < 2048; kc += 32) {
    const int base = kc + asel;
    v16h a;
#pragma unroll
    for (int i = 0; i < 8; ++i) {
      a[i]     = (_Float16)Sa[base + i];
      a[i + 8] = (_Float16)Sa[base + 16 + i];
    }
    acc = WMMA_F16F32(a, load_b16(vrow, kc + bsel), acc);
  }
#pragma unroll
  for (int g = 0; g < 8; ++g) {
    const int n = nt * 16 + g + ((lane >> 4) << 3);
    const int d = dt + (lane & 15);
    aoh[((size_t)b * 2048 + n) * 768 + h * 64 + d] = (_Float16)acc[g];
  }
}

// ---------------------------------------------------------------------------
// Stage 3: out = attn_out @ proj_w^T + proj_b  (f32 output).
__global__ void proj_gemm(const _Float16* __restrict__ aoh,
                          const _Float16* __restrict__ wph,
                          const float* __restrict__ bias,
                          float* __restrict__ out) {
  const int lane = threadIdx.x & 31;
  const int wave = threadIdx.x >> 5;
  const int tile = blockIdx.x * 4 + wave;     // 256*48 = 12288 tiles
  const int NT = 768 / 16;                    // 48
  const int mt = tile / NT;
  const int nt = tile % NT;
  const int asel = (lane >> 4) << 3;
  const int bsel = (lane >> 4) << 3;

  const _Float16* arow = aoh + (mt * 16 + (lane & 15)) * 768;
  const _Float16* brow = wph + (nt * 16 + (lane & 15)) * 768;

  v8f c = {};
  for (int kc = 0; kc < 768; kc += 32) {
    v16h a = load_a16(arow, kc + asel);
    v16h b = load_b16(brow, kc + bsel);
    c = WMMA_F16F32(a, b, c);
  }
#pragma unroll
  for (int g = 0; g < 8; ++g) {
    const int r  = mt * 16 + g + ((lane >> 4) << 3);
    const int cc = nt * 16 + (lane & 15);
    out[r * 768 + cc] = c[g] + bias[cc];
  }
}

// ---------------------------------------------------------------------------
extern "C" void kernel_launch(void* const* d_in, const int* in_sizes, int n_in,
                              void* d_out, int out_size, void* d_ws, size_t ws_size,
                              hipStream_t stream) {
  const float* x      = (const float*)d_in[0];   // [2,2048,768]
  const float* qkv_w  = (const float*)d_in[1];   // [2304,768]
  const float* proj_w = (const float*)d_in[2];   // [768,768]
  const float* proj_b = (const float*)d_in[3];   // [768]
  const float* temp   = (const float*)d_in[4];   // [1]
  const int*   topk   = (const int*)d_in[5];     // scalar

  char* ws = (char*)d_ws;
  // Workspace layout (bytes):
  _Float16* xh    = (_Float16*)(ws + 0);          // 4096*768   f16 =  6291456
  _Float16* wqkvh = (_Float16*)(ws + 6291456);    // 2304*768   f16 =  3538944
  _Float16* wprjh = (_Float16*)(ws + 9830400);    // 768*768    f16 =  1179648
  _Float16* qh    = (_Float16*)(ws + 11010048);   // 24*2048*64 f16 =  6291456
  _Float16* kh    = (_Float16*)(ws + 17301504);   // 24*2048*64 f16 =  6291456
  _Float16* vth   = (_Float16*)(ws + 23592960);   // 24*64*2048 f16 =  6291456
  _Float16* aoh   = (_Float16*)(ws + 29884416);   // 4096*768   f16 =  6291456
  // total: 36175872 bytes

  // Stage 0: f32 -> f16 conversions.
  {
    const int nx = 2 * 2048 * 768;
    cvt_f32_to_f16<<<(nx + 255) / 256, 256, 0, stream>>>(x, xh, nx);
    const int nw = 2304 * 768;
    cvt_f32_to_f16<<<(nw + 255) / 256, 256, 0, stream>>>(qkv_w, wqkvh, nw);
    const int np = 768 * 768;
    cvt_f32_to_f16<<<(np + 255) / 256, 256, 0, stream>>>(proj_w, wprjh, np);
  }

  // Stage 1: QKV projection. 36864 wave-tiles / 4 per block = 9216 blocks.
  qkv_gemm<<<9216, 128, 0, stream>>>(xh, wqkvh, qh, kh, vth);

  // Stage 2: fused filtered attention. B*H*(N/16) = 3072 blocks; 128KB LDS.
  attn_fused<<<3072, 128, 16 * 2048 * sizeof(float), stream>>>(qh, kh, vth, aoh, temp, topk);

  // Stage 3: output projection. 12288 wave-tiles / 4 = 3072 blocks.
  proj_gemm<<<3072, 128, 0, stream>>>(aoh, wprjh, proj_b, (float*)d_out);
}
